// KalmanNetSophisticated_79628693668164
// MI455X (gfx1250) — matrix operand order repaired and checked by reading, not compiled, not run on hardware
//
#include <hip/hip_runtime.h>

// ---------------------------------------------------------------------------
// CDNA5 (gfx1250) KalmanNet: bf16 WMMA mat-vec engine + small fused kernels.
// Weights converted fp32->bf16 once per launch (deterministic, re-done every
// call). Mat-vec via v_wmma_f32_16x16x32_bf16: A = 16 weight rows, B = x
// broadcast into all 16 columns; accumulate over K in v8f.
// Two specialized inner loops (LDS-staged x vs. global x) so the B-operand
// loads lower to ds_load_b128 / global_load_b128 instead of flat_load.
// ---------------------------------------------------------------------------

typedef __attribute__((ext_vector_type(16))) __bf16       v16bf;
typedef __attribute__((ext_vector_type(8)))  float        v8f;
typedef __attribute__((ext_vector_type(4)))  unsigned int u32x4;

union ABPack { v16bf v; u32x4 q[2]; };

__device__ __forceinline__ unsigned short f2bf(float f) {
    unsigned int u = __float_as_uint(f);
    u += 0x7FFFu + ((u >> 16) & 1u);          // round-to-nearest-even
    return (unsigned short)(u >> 16);
}

__device__ __forceinline__ float sigmoidf(float x) {
    return 1.0f / (1.0f + __expf(-x));
}

// ---------------------------------------------------------------- conversion
__global__ void f32_to_bf16_kernel(const float* __restrict__ src,
                                   unsigned short* __restrict__ dst, int n) {
    int stride = (int)(blockDim.x * gridDim.x);
    for (int i = (int)(blockIdx.x * blockDim.x + threadIdx.x); i < n; i += stride)
        dst[i] = f2bf(src[i]);
}

// ------------------------------------------------------------- WMMA mat-vec
// out[rows] = act( W[rows x cols](bf16) @ x[cols](bf16) + bias )
// One wave per 16-row tile. A layout (16-bit, 16x32): lanes 0-15 = rows,
// lane L holds K {kc..kc+7, kc+16..kc+23}; lane L+16 holds {kc+8..15,
// kc+24..31}. B: same K order as A (bcast over N): lanes 0-15 hold K
// {kc..kc+7, kc+16..kc+23}, lanes 16-31 hold K {kc+8..15, kc+24..31}. D: lane 0 -> rows r0..r0+7, lane 16 -> r0+8..r0+15.
__launch_bounds__(128)
__global__ void matvec_wmma_kernel(const unsigned short* __restrict__ W,
                                   const unsigned short* __restrict__ x,
                                   const float* __restrict__ bias,
                                   float* __restrict__ outf,
                                   unsigned short* __restrict__ outbf,
                                   int rows, int cols, int relu)
{
    __shared__ __align__(32) unsigned short xs[2048];
    const int ntiles  = rows >> 4;
    const int wave    = (int)blockIdx.x * ((int)blockDim.x >> 5) + ((int)threadIdx.x >> 5);
    const int lane    = (int)threadIdx.x & 31;
    const bool active = wave < ntiles;
    const bool staged = cols <= 2048;

    if (staged) {                              // stage x through LDS
        for (int i = (int)threadIdx.x; i < cols; i += (int)blockDim.x) xs[i] = x[i];
        __syncthreads();
    }
    if (!active) return;                       // wave-uniform exit

    const int r0     = wave << 4;
    const int half   = lane >> 4;              // 0 or 1
    const int aShift = half << 3;              // in halves
    const int bShift = half << 3;
    const unsigned short* wrow = W + (size_t)(r0 + (lane & 15)) * (size_t)cols + aShift;

    v8f acc = {0.f, 0.f, 0.f, 0.f, 0.f, 0.f, 0.f, 0.f};

    if (staged) {
        // B from LDS -> ds_load_b128
        const unsigned short* xl = xs + bShift;     // addrspace(3) known
        for (int kc = 0; kc < cols; kc += 32) {
            ABPack a, b;
            a.q[0] = *(const u32x4*)(wrow + kc);
            a.q[1] = *(const u32x4*)(wrow + kc + 16);
            b.q[0] = *(const u32x4*)(xl + kc);
            b.q[1] = *(const u32x4*)(xl + kc + 16);
            __builtin_prefetch(wrow + kc + 256, 0, 1);
            acc = __builtin_amdgcn_wmma_f32_16x16x32_bf16(false, a.v, false, b.v,
                                                          (short)0, acc, false, false);
        }
    } else {
        // B from global -> global_load_b128
        const unsigned short* xg = x + bShift;
        for (int kc = 0; kc < cols; kc += 32) {
            ABPack a, b;
            a.q[0] = *(const u32x4*)(wrow + kc);
            a.q[1] = *(const u32x4*)(wrow + kc + 16);
            b.q[0] = *(const u32x4*)(xg + kc);
            b.q[1] = *(const u32x4*)(xg + kc + 16);
            __builtin_prefetch(wrow + kc + 256, 0, 1);
            acc = __builtin_amdgcn_wmma_f32_16x16x32_bf16(false, a.v, false, b.v,
                                                          (short)0, acc, false, false);
        }
    }

    if ((lane & 15) == 0) {                    // lanes 0 and 16 own the outputs
        const int rb = r0 + (half << 3);
        #pragma unroll
        for (int r = 0; r < 8; ++r) {
            float v = acc[r];
            if (bias)  v += bias[rb + r];
            if (relu)  v = fmaxf(v, 0.0f);
            if (outf)  outf[rb + r]  = v;
            if (outbf) outbf[rb + r] = f2bf(v);
        }
    }
}

// ----------------------------------------------------------- GRU elementwise
__global__ void gru_combine_kernel(const float* __restrict__ gi,
                                   const float* __restrict__ gh,
                                   const float* __restrict__ bih,
                                   const float* __restrict__ bhh,
                                   const float* __restrict__ hin,
                                   float* __restrict__ hout,
                                   unsigned short* __restrict__ bf0,
                                   unsigned short* __restrict__ bf1, int dh)
{
    int i = (int)(blockIdx.x * blockDim.x + threadIdx.x);
    if (i >= dh) return;
    float r = sigmoidf(gi[i]        + bih[i]        + gh[i]        + bhh[i]);
    float z = sigmoidf(gi[dh + i]   + bih[dh + i]   + gh[dh + i]   + bhh[dh + i]);
    float n = tanhf   (gi[2*dh + i] + bih[2*dh + i] + r * (gh[2*dh + i] + bhh[2*dh + i]));
    float h = (1.0f - z) * n + z * hin[i];
    hout[i] = h;
    unsigned short hb = f2bf(h);
    if (bf0) bf0[i] = hb;
    if (bf1) bf1[i] = hb;
}

// ------------------------------------------------------- per-step prediction
__global__ void k_predict_kernel(const float* __restrict__ F,
                                 const float* __restrict__ xpost,
                                 const float* __restrict__ xprev,
                                 const float* __restrict__ xprpr,
                                 const float* __restrict__ yprev,
                                 const float* __restrict__ obs_t,
                                 const float* __restrict__ Hm,
                                 const float* __restrict__ w5, const float* __restrict__ b5,
                                 const float* __restrict__ w6, const float* __restrict__ b6,
                                 const float* __restrict__ w7, const float* __restrict__ b7,
                                 float* __restrict__ xprior, float* __restrict__ yprior,
                                 unsigned short* __restrict__ out5_bf,
                                 unsigned short* __restrict__ out6_bf,
                                 unsigned short* __restrict__ out7_bf)
{
    __shared__ float sxp[32], sxpr[32], fe[32], fu[32], oc[32], nrm[4];
    int tid = (int)threadIdx.x;
    if (tid < 32) sxp[tid] = xpost[tid];
    __syncthreads();
    if (tid < 32) {                                    // x_prior = F @ x_post
        float a = 0.f;
        for (int k = 0; k < 32; ++k) a += F[tid * 32 + k] * sxp[k];
        sxpr[tid] = a; xprior[tid] = a;
    }
    __syncthreads();
    if (tid < 16) {                                    // y_prior = H @ x_prior
        float a = 0.f;
        for (int k = 0; k < 32; ++k) a += Hm[tid * 32 + k] * sxpr[k];
        yprior[tid] = a;
        float y = obs_t[tid];
        oc[tid]      = y - yprev[tid];                 // obs_diff (raw)
        oc[16 + tid] = y - a;                          // obs_innov (raw)
    }
    if (tid < 32) {
        fe[tid] = sxp[tid] - xprev[tid];               // fw_evol (raw)
        fu[tid] = sxp[tid] - xprpr[tid];               // fw_upd  (raw)
    }
    __syncthreads();
    if (tid < 4) {                                     // l2 norms
        const float* v = (tid == 0) ? fe : (tid == 1) ? fu : (tid == 2) ? oc : oc + 16;
        int n = (tid < 2) ? 32 : 16;
        float s = 0.f;
        for (int k = 0; k < n; ++k) s += v[k] * v[k];
        nrm[tid] = fmaxf(sqrtf(s), 1e-12f);
    }
    __syncthreads();
    if (tid < 32) { fe[tid] /= nrm[0]; fu[tid] /= nrm[1]; }
    if (tid < 16) { oc[tid] /= nrm[2]; oc[16 + tid] /= nrm[3]; }
    __syncthreads();
    if (tid < 160) {                                   // fc5 / fc6 / fc7 (+relu)
        float a5 = b5[tid], a6 = b6[tid], a7 = b7[tid];
        for (int k = 0; k < 32; ++k) {
            a5 += w5[tid * 32 + k] * fe[k];
            a6 += w6[tid * 32 + k] * fu[k];
            a7 += w7[tid * 32 + k] * oc[k];
        }
        out5_bf[tid] = f2bf(fmaxf(a5, 0.f));
        out6_bf[tid] = f2bf(fmaxf(a6, 0.f));
        out7_bf[tid] = f2bf(fmaxf(a7, 0.f));
    }
}

// ----------------------------------------------------------- per-step update
__global__ void k_update_kernel(const float* __restrict__ K512,
                                const float* __restrict__ obs_t,
                                const float* __restrict__ yprior,
                                const float* __restrict__ xprior,
                                float* __restrict__ xpost,
                                float* __restrict__ xprev,
                                float* __restrict__ xprpr,
                                float* __restrict__ yprev,
                                float* __restrict__ traj_t)
{
    __shared__ float dy[16];
    int tid = (int)threadIdx.x;
    if (tid < 16) dy[tid] = obs_t[tid] - yprior[tid];
    __syncthreads();
    if (tid < 32) {
        float old_post = xpost[tid];
        float xp = xprior[tid];
        float a = xp;
        for (int j = 0; j < 16; ++j) a += K512[tid * 16 + j] * dy[j];
        xprev[tid]  = old_post;                        // x_prev_post  <- x_post
        xprpr[tid]  = xp;                              // x_prior_prev <- x_prior
        xpost[tid]  = a;                               // x_post       <- x_new
        traj_t[tid] = a;
    }
    if (tid < 16) yprev[tid] = obs_t[tid];
}

// ------------------------------------------------------------------ initialize
__global__ void k_init_kernel(const float* __restrict__ A,
                              const float* __restrict__ x0,
                              const float* __restrict__ hq0,
                              const float* __restrict__ hsg0,
                              const float* __restrict__ hs0,
                              const float* __restrict__ Hm,
                              float* __restrict__ F,
                              float* __restrict__ xpost, float* __restrict__ xprev,
                              float* __restrict__ xprpr, float* __restrict__ yprev,
                              float* __restrict__ hqf, float* __restrict__ hsgf,
                              float* __restrict__ hsf,
                              unsigned short* __restrict__ hq_bf,
                              unsigned short* __restrict__ hsg_bf0,
                              unsigned short* __restrict__ hsg_bf1,
                              unsigned short* __restrict__ hs_bf0,
                              unsigned short* __restrict__ hs_bf1)
{
    int tid = (int)threadIdx.x;                        // 1024 threads
    {                                                  // F = I + dtA + (dtA)^2/2
        int i = tid >> 5, j = tid & 31;
        float s = 0.f;
        for (int k = 0; k < 32; ++k) s += (0.1f * A[i * 32 + k]) * (0.1f * A[k * 32 + j]);
        float f = 0.1f * A[i * 32 + j] + 0.5f * s;
        if (i == j) f += 1.0f;
        F[tid] = f;
    }
    if (tid < 32) { xpost[tid] = x0[tid]; xprev[tid] = x0[tid]; xprpr[tid] = x0[tid]; }
    if (tid < 16) {
        float a = 0.f;
        for (int k = 0; k < 32; ++k) a += Hm[tid * 32 + k] * x0[k];
        yprev[tid] = a;                                // y_prev0 = H @ x0
    }
    {
        float v = hq0[tid];  hqf[tid]  = v; hq_bf[tid] = f2bf(v);
        float w = hsg0[tid]; hsgf[tid] = w;
        unsigned short wb = f2bf(w); hsg_bf0[tid] = wb; hsg_bf1[tid] = wb;
    }
    if (tid < 256) {
        float v = hs0[tid]; hsf[tid] = v;
        unsigned short vb = f2bf(v); hs_bf0[tid] = vb; hs_bf1[tid] = vb;
    }
}

// ---------------------------------------------------------------------------
extern "C" void kernel_launch(void* const* d_in, const int* in_sizes, int n_in,
                              void* d_out, int out_size, void* d_ws, size_t ws_size,
                              hipStream_t stream)
{
    (void)in_sizes; (void)n_in; (void)out_size; (void)ws_size;
    const float* obs    = (const float*)d_in[0];
    const float* x0     = (const float*)d_in[1];
    const float* hq0    = (const float*)d_in[2];
    const float* hsg0   = (const float*)d_in[3];
    const float* hs0    = (const float*)d_in[4];
    const float* Hm     = (const float*)d_in[5];
    const float* Am     = (const float*)d_in[6];
    const float* gq_wih = (const float*)d_in[7];
    const float* gq_whh = (const float*)d_in[8];
    const float* gq_bih = (const float*)d_in[9];
    const float* gq_bhh = (const float*)d_in[10];
    const float* gg_wih = (const float*)d_in[11];
    const float* gg_whh = (const float*)d_in[12];
    const float* gg_bih = (const float*)d_in[13];
    const float* gg_bhh = (const float*)d_in[14];
    const float* gs_wih = (const float*)d_in[15];
    const float* gs_whh = (const float*)d_in[16];
    const float* gs_bih = (const float*)d_in[17];
    const float* gs_bhh = (const float*)d_in[18];
    const float* fc1_w  = (const float*)d_in[19];
    const float* fc1_b  = (const float*)d_in[20];
    const float* fc2a_w = (const float*)d_in[21];
    const float* fc2a_b = (const float*)d_in[22];
    const float* fc2b_w = (const float*)d_in[23];
    const float* fc2b_b = (const float*)d_in[24];
    const float* fc3_w  = (const float*)d_in[25];
    const float* fc3_b  = (const float*)d_in[26];
    const float* fc4_w  = (const float*)d_in[27];
    const float* fc4_b  = (const float*)d_in[28];
    const float* fc5_w  = (const float*)d_in[29];
    const float* fc5_b  = (const float*)d_in[30];
    const float* fc6_w  = (const float*)d_in[31];
    const float* fc6_b  = (const float*)d_in[32];
    const float* fc7_w  = (const float*)d_in[33];
    const float* fc7_b  = (const float*)d_in[34];
    float* traj = (float*)d_out;

    // ---- workspace bump allocator (256B aligned)
    char* wp = (char*)d_ws;
    auto alloc_us = [&](size_t n) {
        unsigned short* r = (unsigned short*)wp;
        wp += ((n * 2 + 255) & ~(size_t)255);
        return r;
    };
    auto alloc_f = [&](size_t n) {
        float* r = (float*)wp;
        wp += ((n * 4 + 255) & ~(size_t)255);
        return r;
    };

    // bf16 weight images (~212 MB)
    unsigned short* bW_gq_ih = alloc_us((size_t)3072 * 160);
    unsigned short* bW_gq_hh = alloc_us((size_t)3072 * 1024);
    unsigned short* bW_gg_ih = alloc_us((size_t)3072 * 1184);
    unsigned short* bW_gg_hh = alloc_us((size_t)3072 * 1024);
    unsigned short* bW_gs_ih = alloc_us((size_t)768 * 416);
    unsigned short* bW_gs_hh = alloc_us((size_t)768 * 256);
    unsigned short* bW_fc1   = alloc_us((size_t)256 * 1024);
    unsigned short* bW_fc2a  = alloc_us((size_t)51200 * 1280);
    unsigned short* bW_fc2b  = alloc_us((size_t)512 * 51200);
    unsigned short* bW_fc3   = alloc_us((size_t)1024 * 768);
    unsigned short* bW_fc4   = alloc_us((size_t)1024 * 2048);

    // bf16 activation / concat buffers (aliased layouts)
    unsigned short* xq_in   = alloc_us(160);    // out5
    unsigned short* sig_in  = alloc_us(1184);   // [hq(1024) | out6(160)]
    unsigned short* fc2a_in = alloc_us(1280);   // [hsig(1024) | hs(256)]
    unsigned short* fc3_in  = alloc_us(768);    // [hs(256) | out2(512)]
    unsigned short* fc4_in  = alloc_us(2048);   // [hsig_pre_fc4(1024) | out3(1024)]
    unsigned short* s_in    = alloc_us(416);    // [out1(256) | out7(160)]
    unsigned short* out2a   = alloc_us(51200);  // relu(fc2a)

    // fp32 state / gates
    float* Fmat   = alloc_f(1024);
    float* xpost  = alloc_f(32);
    float* xprev  = alloc_f(32);
    float* xprpr  = alloc_f(32);
    float* yprev  = alloc_f(16);
    float* xprior = alloc_f(32);
    float* yprior = alloc_f(16);
    float* gi_q   = alloc_f(3072);
    float* gh_q   = alloc_f(3072);
    float* gi_g   = alloc_f(3072);
    float* gh_g   = alloc_f(3072);
    float* gi_s   = alloc_f(768);
    float* gh_s   = alloc_f(768);
    float* hq_f   = alloc_f(1024);
    float* hsg_f  = alloc_f(1024);
    float* hs_f   = alloc_f(256);
    float* out2_f = alloc_f(512);

    // ---- one-time (per launch) weight conversion
    auto conv = [&](const float* s, unsigned short* d, size_t n) {
        size_t b = (n + 255) / 256; if (b > 4096) b = 4096;
        f32_to_bf16_kernel<<<dim3((unsigned)b), dim3(256), 0, stream>>>(s, d, (int)n);
    };
    conv(gq_wih, bW_gq_ih, (size_t)3072 * 160);
    conv(gq_whh, bW_gq_hh, (size_t)3072 * 1024);
    conv(gg_wih, bW_gg_ih, (size_t)3072 * 1184);
    conv(gg_whh, bW_gg_hh, (size_t)3072 * 1024);
    conv(gs_wih, bW_gs_ih, (size_t)768 * 416);
    conv(gs_whh, bW_gs_hh, (size_t)768 * 256);
    conv(fc1_w,  bW_fc1,   (size_t)256 * 1024);
    conv(fc2a_w, bW_fc2a,  (size_t)51200 * 1280);
    conv(fc2b_w, bW_fc2b,  (size_t)512 * 51200);
    conv(fc3_w,  bW_fc3,   (size_t)1024 * 768);
    conv(fc4_w,  bW_fc4,   (size_t)1024 * 2048);

    k_init_kernel<<<1, 1024, 0, stream>>>(Am, x0, hq0, hsg0, hs0, Hm,
        Fmat, xpost, xprev, xprpr, yprev, hq_f, hsg_f, hs_f,
        /*hq_bf=*/sig_in, /*hsg_bf0=*/fc2a_in, /*hsg_bf1=*/fc4_in,
        /*hs_bf0=*/fc3_in, /*hs_bf1=*/fc2a_in + 1024);

    auto mv = [&](const unsigned short* W, const unsigned short* xv, const float* bias,
                  float* outf, unsigned short* outbf, int rows, int cols, int relu) {
        int tiles  = rows >> 4;
        int blocks = (tiles + 3) >> 2;                 // 4 waves / block
        matvec_wmma_kernel<<<dim3(blocks), dim3(128), 0, stream>>>(
            W, xv, bias, outf, outbf, rows, cols, relu);
    };
    auto gcomb = [&](const float* gi, const float* gh, const float* bih, const float* bhh,
                     float* hf, unsigned short* b0, unsigned short* b1, int dh) {
        gru_combine_kernel<<<dim3((dh + 255) / 256), dim3(256), 0, stream>>>(
            gi, gh, bih, bhh, hf, hf, b0, b1, dh);
    };

    for (int t = 0; t < 200; ++t) {
        const float* yt = obs + (size_t)t * 16;
        // prediction + features + fc5/fc6/fc7
        k_predict_kernel<<<1, 256, 0, stream>>>(Fmat, xpost, xprev, xprpr, yprev, yt, Hm,
            fc5_w, fc5_b, fc6_w, fc6_b, fc7_w, fc7_b,
            xprior, yprior, xq_in, sig_in + 1024, s_in + 256);
        // gru_q
        mv(bW_gq_ih, xq_in,  nullptr, gi_q, nullptr, 3072, 160,  0);
        mv(bW_gq_hh, sig_in, nullptr, gh_q, nullptr, 3072, 1024, 0);   // h = old hq
        gcomb(gi_q, gh_q, gq_bih, gq_bhh, hq_f, sig_in, nullptr, 1024);
        // gru_sigma: x = [hq | out6], h = old hsig (fc2a_in[0:1024])
        mv(bW_gg_ih, sig_in,  nullptr, gi_g, nullptr, 3072, 1184, 0);
        mv(bW_gg_hh, fc2a_in, nullptr, gh_g, nullptr, 3072, 1024, 0);
        gcomb(gi_g, gh_g, gg_bih, gg_bhh, hsg_f, fc2a_in, fc4_in, 1024);
        // fc1 -> out1 into s_in[0:256]
        mv(bW_fc1, fc2a_in, fc1_b, nullptr, s_in, 256, 1024, 1);
        // gru_s: x = [out1 | out7], h = old hs (fc3_in[0:256])
        mv(bW_gs_ih, s_in,   nullptr, gi_s, nullptr, 768, 416, 0);
        mv(bW_gs_hh, fc3_in, nullptr, gh_s, nullptr, 768, 256, 0);
        gcomb(gi_s, gh_s, gs_bih, gs_bhh, hs_f, fc3_in, fc2a_in + 1024, 256);
        // fc2a (relu) -> fc2b -> out2 (K matrix)
        mv(bW_fc2a, fc2a_in, fc2a_b, nullptr, out2a, 51200, 1280, 1);
        mv(bW_fc2b, out2a, fc2b_b, out2_f, fc3_in + 256, 512, 51200, 0);
        // fc3 (relu) -> fc4 (relu) overwrites hsig
        mv(bW_fc3, fc3_in, fc3_b, nullptr, fc4_in + 1024, 1024, 768, 1);
        mv(bW_fc4, fc4_in, fc4_b, hsg_f, fc2a_in, 1024, 2048, 1);
        // Kalman update + carry rotation
        k_update_kernel<<<1, 64, 0, stream>>>(out2_f, yt, yprior, xprior,
            xpost, xprev, xprpr, yprev, traj + (size_t)t * 32);
    }
}
